// DynamicNeuralBasis_22746146799964
// MI455X (gfx1250) — matrix operand order repaired and checked by reading, not compiled, not run on hardware
//
#include <hip/hip_runtime.h>

// ---------------------------------------------------------------------------
// Galerkin linear attention for MI455X (gfx1250, wave32, WMMA bf16).
//   B=4, N=8192, XD=256, F=512, H=8, D=64, EPS=1e-6
// Pipeline:
//   prep_weights : W^T -> bf16 (exp(log_scale) folded into V columns)
//   proj_ln      : P = X @ W (WMMA bf16), weight tiles streamed by the
//                  Tensor Data Mover (tensor_load_to_lds, double-buffered,
//                  s_wait_tensorcnt), fused per-64-group LayerNorm
//   ctx_kernel   : per (b,h): partial ctx = Kh^T @ Vh   (WMMA bf16)
//   out_kernel   : out = Qh @ (sum of ctx partials) / N (WMMA bf16, fp32 out)
// ---------------------------------------------------------------------------

#define BB   4
#define NN   8192
#define XD   256
#define FF   512
#define HH   8
#define DD   64
#define EPSL 1e-6f

typedef __bf16        v16bf __attribute__((ext_vector_type(16)));
typedef __bf16        v2bf  __attribute__((ext_vector_type(2)));
typedef float         v8f   __attribute__((ext_vector_type(8)));
typedef unsigned int  v4u   __attribute__((ext_vector_type(4)));
typedef unsigned int  u32x4 __attribute__((ext_vector_type(4)));
typedef int           i32x4 __attribute__((ext_vector_type(4)));
typedef int           i32x8 __attribute__((ext_vector_type(8)));

#if __has_builtin(__builtin_amdgcn_tensor_load_to_lds) && \
    __has_builtin(__builtin_amdgcn_s_wait_tensorcnt)
#define USE_TDM 1
#endif

union Frag {                 // 16 bf16 operand fragment (8 VGPRs)
    v16bf        bf;
    unsigned int w[8];
};
union AccU {                 // 16x16 fp32 accumulator tile (8 VGPRs)
    v8f   v;
    float f[8];
};

__device__ __forceinline__ unsigned short f2bf(float f) {
    unsigned u = __builtin_bit_cast(unsigned, f);
    unsigned r = (u + 0x7FFFu + ((u >> 16) & 1u)) >> 16;   // RNE
    return (unsigned short)r;
}
__device__ __forceinline__ unsigned pack2(unsigned short lo, unsigned short hi) {
    return (unsigned)lo | ((unsigned)hi << 16);
}
__device__ __forceinline__ unsigned pk2bf(float a, float b) {
#if __has_builtin(__builtin_amdgcn_cvt_pk_bf16_f32)
    v2bf h = __builtin_amdgcn_cvt_pk_bf16_f32(a, b);       // v_cvt_pk_bf16_f32
    return __builtin_bit_cast(unsigned, h);
#else
    return pack2(f2bf(a), f2bf(b));
#endif
}
__device__ __forceinline__ v8f wmma_bf16(const Frag& a, const Frag& b, v8f c) {
    // D = A(16x32 bf16) * B(32x16 bf16) + C(16x16 f32)
    return __builtin_amdgcn_wmma_f32_16x16x32_bf16(
        false, a.bf, false, b.bf, (short)0, c, false, false);
}

#ifdef USE_TDM
// TDM load of a [tile_dim1=128 rows x tile_dim0=32 elems] bf16 tile from a
// row-major tensor with row stride 256 elems into contiguous LDS.
// D# layout per CDNA5 ISA ch.8 (group0: count/lds/global addr/type,
// group1: data_size, dims, tile dims, strides). Groups 2/3 zero (<=2D).
__device__ __forceinline__ void tdm_load_wtile(const unsigned short* gsrc,
                                               unsigned lds_byte_off) {
    unsigned long long ga = (unsigned long long)(__SIZE_TYPE__)gsrc;
    u32x4 g0 = {
        0x1u,                                   // count=1, user descriptor
        lds_byte_off,                           // lds_addr (bytes)
        (unsigned)(ga & 0xFFFFFFFFull),         // global_addr[31:0]
        (unsigned)((ga >> 32) & 0x1FFFFFFull) | (2u << 30)  // addr hi | type=2
    };
    i32x8 g1 = {
        (int)(1u << 16),            // workgroup_mask=0, data_size=1 (2 bytes)
        (int)(256u << 16),          // tensor_dim0 = 256 (low 16 @ bits 63:48)
        (int)(512u << 16),          // tensor_dim1 = 512 (low 16 @ bits 95:80)
        (int)(32u << 16),           // tile_dim0   = 32  (bits 127:112)
        (int)128,                   // tile_dim1   = 128, tile_dim2 = 0
        (int)256,                   // tensor_dim0_stride = 256
        0, 0                        // stride hi / tensor_dim1_stride = 0
    };
    i32x4 gz = { 0, 0, 0, 0 };
#if __clang_major__ >= 23
    i32x8 gz8 = { 0, 0, 0, 0, 0, 0, 0, 0 };
    __builtin_amdgcn_tensor_load_to_lds(g0, g1, gz, gz, gz8, 0);
#else
    __builtin_amdgcn_tensor_load_to_lds(g0, g1, gz, gz, 0);
#endif
}
__device__ __forceinline__ unsigned lds_offset_of(const void* p) {
    return (unsigned)(__SIZE_TYPE__)(__attribute__((address_space(3))) const void*)p;
}
#endif

// ---------------------------------------------------------------------------
// Kernel 0: transposed bf16 weights WT[f][k] = W[k][f]; V scaled by exp(ls[f])
// ---------------------------------------------------------------------------
__global__ __launch_bounds__(256) void prep_weights(
    const float* __restrict__ Qw, const float* __restrict__ Kw,
    const float* __restrict__ Vw, const float* __restrict__ ls,
    unsigned short* __restrict__ WTq, unsigned short* __restrict__ WTk,
    unsigned short* __restrict__ WTv) {
    int idx = blockIdx.x * 256 + threadIdx.x;   // 0 .. F*XD-1
    int f = idx >> 8;                           // 0..511
    int k = idx & 255;                          // 0..255
    float s = __expf(ls[f]);
    WTq[idx] = f2bf(Qw[k * FF + f]);
    WTk[idx] = f2bf(Kw[k * FF + f]);
    WTv[idx] = f2bf(Vw[k * FF + f] * s);
}

// ---------------------------------------------------------------------------
// Kernel 1: projection GEMM + fused per-64-group LayerNorm.
// Grid: (N/128, F/128, B), block 256 (8 waves). Wave tile: 32 rows x 64 cols
// (one full LN group per row per wave). Weight tiles arrive via TDM.
// ---------------------------------------------------------------------------
__global__ __launch_bounds__(256) void proj_ln(
    const float* __restrict__ X,            // [B, N, XD] fp32
    const unsigned short* __restrict__ WT,  // [F, XD] bf16 (transposed weights)
    const float* __restrict__ lnw, const float* __restrict__ lnb,  // [64]
    unsigned short* __restrict__ Pn) {      // [B, N, F] normalized bf16
    const int rb = blockIdx.x;              // row block (128 rows)
    const int cb = blockIdx.y;              // col block (128 cols)
    const int b  = blockIdx.z;

    const float* Xb = X + (size_t)b * NN * XD;
    unsigned short* Out = Pn + (size_t)b * NN * FF;
    const unsigned short* Wblk = WT + (size_t)(cb * 128) * XD;

    __shared__ unsigned short sA[128 * 32];     // [row][k]        8KB bf16
    __shared__ unsigned short sB[2][128 * 32];  // [col][k] x2    16KB bf16

    const int tid  = threadIdx.x;
    const int wid  = tid >> 5;
    const int lane = tid & 31;
    const int hlf  = lane >> 4;             // which 16-lane half
    const int l16  = lane & 15;
    const int wr   = (wid >> 1) * 32;       // wave row offset: 0/32/64/96
    const int wc   = (wid & 1) * 64;        // wave col offset: 0/64

    AccU acc[2][4] = {};
    int cur = 0;

#ifdef USE_TDM
    const unsigned ldsB[2] = { lds_offset_of(&sB[0][0]), lds_offset_of(&sB[1][0]) };
    if (tid == 0) tdm_load_wtile(Wblk, ldsB[0]);   // prologue DMA, buffer 0
#endif

    for (int kk = 0; kk < XD; kk += 32) {
        // ---- stage A: 128x32 fp32 -> bf16 into LDS (convert on the fly) ----
        #pragma unroll
        for (int it = 0; it < 4; ++it) {
            int idx = tid + it * 256;       // 1024 float4 chunks
            int r   = idx >> 3;
            int c4  = (idx & 7) * 4;
            float4 v = *(const float4*)&Xb[(size_t)(rb * 128 + r) * XD + kk + c4];
            *(unsigned*)&sA[r * 32 + c4]     = pk2bf(v.x, v.y);
            *(unsigned*)&sA[r * 32 + c4 + 2] = pk2bf(v.z, v.w);
        }
#ifdef USE_TDM
        // ---- stage B: Tensor Data Mover, double-buffered ----
        if (tid == 0) {
            if (kk + 32 < XD) {
                tdm_load_wtile(Wblk + kk + 32, ldsB[cur ^ 1]);  // prefetch next
                __builtin_amdgcn_s_wait_tensorcnt(1);           // current ready
            } else {
                __builtin_amdgcn_s_wait_tensorcnt(0);
            }
        }
#else
        // ---- stage B (fallback): manual 128 cols x 32 k bf16 into LDS ----
        #pragma unroll
        for (int it = 0; it < 2; ++it) {
            int idx = tid + it * 256;       // 512 x 16B chunks
            int c   = idx >> 2;
            int c8  = (idx & 3) * 8;
            *(v4u*)&sB[cur][c * 32 + c8] = *(const v4u*)&Wblk[(size_t)c * XD + kk + c8];
        }
#endif
        // prefetch next X k-chunk (global_prefetch_b8)
        if (kk + 32 < XD && tid < 128)
            __builtin_prefetch(&Xb[(size_t)(rb * 128 + tid) * XD + kk + 32], 0, 1);
        __syncthreads();

        // ---- fragments + WMMA ----
        Frag afr[2], bfr[4];
        #pragma unroll
        for (int i = 0; i < 2; ++i) {
            int row = wr + 16 * i + l16;    // A: lane holds one row; K halves split
            *(v4u*)&afr[i].w[0] = *(const v4u*)&sA[row * 32 + hlf * 8];
            *(v4u*)&afr[i].w[4] = *(const v4u*)&sA[row * 32 + 16 + hlf * 8];
        }
        #pragma unroll
        for (int j = 0; j < 4; ++j) {
            int col = wc + 16 * j + l16;    // B: lane holds one col; K halves split
            *(v4u*)&bfr[j].w[0] = *(const v4u*)&sB[cur][col * 32 + hlf * 8];
            *(v4u*)&bfr[j].w[4] = *(const v4u*)&sB[cur][col * 32 + 16 + hlf * 8];
        }
        #pragma unroll
        for (int i = 0; i < 2; ++i)
            #pragma unroll
            for (int j = 0; j < 4; ++j)
                acc[i][j].v = wmma_bf16(afr[i], bfr[j], acc[i][j].v);
        __syncthreads();
        cur ^= 1;
    }

    // ---- fused LayerNorm over the wave's 64-col group, then store bf16 ----
    float lw[4], lb[4];
    #pragma unroll
    for (int j = 0; j < 4; ++j) {
        lw[j] = lnw[16 * j + l16];
        lb[j] = lnb[16 * j + l16];
    }
    #pragma unroll
    for (int i = 0; i < 2; ++i) {
        #pragma unroll
        for (int r = 0; r < 8; ++r) {
            float s = 0.f, q = 0.f;
            #pragma unroll
            for (int j = 0; j < 4; ++j) {
                float v = acc[i][j].f[r];
                s += v; q += v * v;
            }
            // butterfly reduce within each 16-lane half (one row per half)
            #pragma unroll
            for (int m = 1; m <= 8; m <<= 1) {
                s += __shfl_xor(s, m, 32);
                q += __shfl_xor(q, m, 32);
            }
            float mu  = s * (1.f / 64.f);
            float var = q * (1.f / 64.f) - mu * mu;
            float inv = rsqrtf(var + EPSL);
            int row = rb * 128 + wr + 16 * i + hlf * 8 + r;
            #pragma unroll
            for (int j = 0; j < 4; ++j) {
                float y = (acc[i][j].f[r] - mu) * inv * lw[j] + lb[j];
                int col = cb * 128 + wc + 16 * j + l16;
                Out[(size_t)row * FF + col] = f2bf(y);
            }
        }
    }
}

// ---------------------------------------------------------------------------
// Kernel 2: per (b,h,segment) partial ctx[64x64] = Kh^T @ Vh over 1024 slices.
// Grid: B*H*8 blocks, 128 threads (4 waves; wave owns one 16-row d-band).
// ---------------------------------------------------------------------------
__global__ __launch_bounds__(128) void ctx_kernel(
    const unsigned short* __restrict__ Kn,  // [B*N*F] normalized bf16
    const unsigned short* __restrict__ Vn,
    float* __restrict__ ctxp) {             // [B*H*8][64*64] partials
    const int blk = blockIdx.x;
    const int seg = blk & 7;
    const int bh  = blk >> 3;               // 0..31

    const unsigned short* Kh = Kn + (size_t)bh * NN * DD;  // [8192 x 64]
    const unsigned short* Vh = Vn + (size_t)bh * NN * DD;
    float* Cp = ctxp + (size_t)blk * (DD * DD);

    __shared__ unsigned short sK[32 * 64];  // [m][d] 4KB
    __shared__ unsigned short sV[32 * 64];  // [m][e] 4KB

    const int tid  = threadIdx.x;
    const int wid  = tid >> 5;              // 0..3 -> d band
    const int lane = tid & 31;
    const int hlf  = lane >> 4;
    const int l16  = lane & 15;

    AccU acc[4] = {};

    const int m_base = seg * 1024;
    for (int m0 = m_base; m0 < m_base + 1024; m0 += 32) {
        #pragma unroll
        for (int it = 0; it < 2; ++it) {
            int idx = tid + it * 128;       // 256 x 16B chunks per matrix
            int m   = idx >> 3;
            int c8  = (idx & 7) * 8;
            *(v4u*)&sK[m * 64 + c8] = *(const v4u*)&Kh[(size_t)(m0 + m) * DD + c8];
            *(v4u*)&sV[m * 64 + c8] = *(const v4u*)&Vh[(size_t)(m0 + m) * DD + c8];
        }
        __syncthreads();

        // A = K^T tile: A[d, m] = K[m, d] (transposed gather from LDS)
        Frag af;
        const int d = wid * 16 + l16;
        #pragma unroll
        for (int v = 0; v < 8; ++v) {
            int mA = (v < 4) ? (hlf * 8 + 2 * v) : (16 + hlf * 8 + 2 * (v - 4));
            af.w[v] = pack2(sK[mA * 64 + d], sK[(mA + 1) * 64 + d]);
        }
        Frag bf4[4];
        #pragma unroll
        for (int j = 0; j < 4; ++j) {
            int e = 16 * j + l16;
            #pragma unroll
            for (int v = 0; v < 8; ++v) {
                int bm = (v < 4) ? (hlf * 8 + 2 * v) : (16 + hlf * 8 + 2 * (v - 4));
                bf4[j].w[v] = pack2(sV[bm * 64 + e], sV[(bm + 1) * 64 + e]);
            }
        }
        #pragma unroll
        for (int j = 0; j < 4; ++j)
            acc[j].v = wmma_bf16(af, bf4[j], acc[j].v);
        __syncthreads();
    }

    const int d0 = 16 * wid + hlf * 8;
    #pragma unroll
    for (int j = 0; j < 4; ++j)
        #pragma unroll
        for (int r = 0; r < 8; ++r)
            Cp[(size_t)(d0 + r) * DD + 16 * j + l16] = acc[j].f[r];
}

// ---------------------------------------------------------------------------
// Kernel 3: out = Qh @ ctx / N. Grid: B*H*32 blocks (256 rows each), 256 thr.
// ctx partials reduced + transposed + (1/N)-scaled into LDS bf16 once per WG.
// ---------------------------------------------------------------------------
__global__ __launch_bounds__(256) void out_kernel(
    const unsigned short* __restrict__ Qn,  // [B*N*F] normalized bf16
    const float* __restrict__ ctxp,         // [B*H*8][64*64]
    float* __restrict__ out) {              // [B, N, F] fp32
    const int blk = blockIdx.x;
    const int rb  = blk & 31;               // 256-row block within head
    const int bh  = blk >> 5;               // 0..31

    const unsigned short* Qh = Qn + (size_t)bh * NN * DD;   // [8192 x 64]
    float* Oh = out + (size_t)bh * NN * DD;
    const float* Cb = ctxp + (size_t)bh * 8 * (DD * DD);

    __shared__ unsigned short sC[64 * 64];  // ctx^T bf16: [e][d], 8KB

    const int tid = threadIdx.x;
    #pragma unroll
    for (int it = 0; it < 16; ++it) {
        int idx = tid + it * 256;           // 4096 elements
        int d = idx >> 6, e = idx & 63;
        float v = 0.f;
        #pragma unroll
        for (int s = 0; s < 8; ++s) v += Cb[(size_t)s * (DD * DD) + idx];
        sC[e * 64 + d] = f2bf(v * (1.f / (float)NN));  // fold 1/seq_len here
    }
    __syncthreads();

    const int wid  = tid >> 5;
    const int lane = tid & 31;
    const int hlf  = lane >> 4;
    const int l16  = lane & 15;
    const int row0 = rb * 256 + wid * 32;

    AccU acc[2][4] = {};
    #pragma unroll
    for (int k0 = 0; k0 < DD; k0 += 32) {
        Frag af[2], bf4[4];
        #pragma unroll
        for (int i = 0; i < 2; ++i) {       // A straight from global (row-major)
            const unsigned short* p = &Qh[(size_t)(row0 + 16 * i + l16) * DD + k0];
            *(v4u*)&af[i].w[0] = *(const v4u*)&p[hlf * 8];
            *(v4u*)&af[i].w[4] = *(const v4u*)&p[16 + hlf * 8];
        }
        #pragma unroll
        for (int j = 0; j < 4; ++j) {       // B from LDS ctx^T (contiguous)
            int e = 16 * j + l16;
            *(v4u*)&bf4[j].w[0] = *(const v4u*)&sC[e * 64 + k0 + hlf * 8];
            *(v4u*)&bf4[j].w[4] = *(const v4u*)&sC[e * 64 + k0 + 16 + hlf * 8];
        }
        #pragma unroll
        for (int i = 0; i < 2; ++i)
            #pragma unroll
            for (int j = 0; j < 4; ++j)
                acc[i][j].v = wmma_bf16(af[i], bf4[j], acc[i][j].v);
    }
    #pragma unroll
    for (int i = 0; i < 2; ++i)
        #pragma unroll
        for (int r = 0; r < 8; ++r) {
            int row = row0 + 16 * i + hlf * 8 + r;
            #pragma unroll
            for (int j = 0; j < 4; ++j)
                Oh[(size_t)row * DD + 16 * j + l16] = acc[i][j].f[r];
        }
}

// ---------------------------------------------------------------------------
extern "C" void kernel_launch(void* const* d_in, const int* in_sizes, int n_in,
                              void* d_out, int out_size, void* d_ws, size_t ws_size,
                              hipStream_t stream) {
    const float* x    = (const float*)d_in[0];
    const float* kv   = (const float*)d_in[1];
    const float* Qw   = (const float*)d_in[2];
    const float* Kw   = (const float*)d_in[3];
    const float* Vw   = (const float*)d_in[4];
    const float* ls   = (const float*)d_in[5];
    const float* lnqw = (const float*)d_in[6];
    const float* lnqb = (const float*)d_in[7];
    const float* lnkw = (const float*)d_in[8];
    const float* lnkb = (const float*)d_in[9];
    const float* lnvw = (const float*)d_in[10];
    const float* lnvb = (const float*)d_in[11];
    float* out = (float*)d_out;

    char* ws = (char*)d_ws;
    const size_t szW  = (size_t)FF * XD * 2;          // 256KB per weight
    const size_t szPn = (size_t)BB * NN * FF * 2;     // 32MB per normalized buf
    unsigned short* WTq = (unsigned short*)(ws);
    unsigned short* WTk = (unsigned short*)(ws + szW);
    unsigned short* WTv = (unsigned short*)(ws + 2 * szW);
    unsigned short* Qn  = (unsigned short*)(ws + 3 * szW);
    unsigned short* Kn  = (unsigned short*)(ws + 3 * szW + szPn);
    unsigned short* Vn  = (unsigned short*)(ws + 3 * szW + 2 * szPn);
    float*          Cp  = (float*)(ws + 3 * szW + 3 * szPn);  // 4MB partials

    prep_weights<<<dim3((FF * XD) / 256), dim3(256), 0, stream>>>(
        Qw, Kw, Vw, ls, WTq, WTk, WTv);

    dim3 gproj(NN / 128, FF / 128, BB);
    proj_ln<<<gproj, dim3(256), 0, stream>>>(x,  WTq, lnqw, lnqb, Qn);
    proj_ln<<<gproj, dim3(256), 0, stream>>>(kv, WTk, lnkw, lnkb, Kn);
    proj_ln<<<gproj, dim3(256), 0, stream>>>(kv, WTv, lnvw, lnvb, Vn);

    ctx_kernel<<<dim3(BB * HH * 8), dim3(128), 0, stream>>>(Kn, Vn, Cp);
    out_kernel<<<dim3(BB * HH * 32), dim3(256), 0, stream>>>(Qn, Cp, out);
}
